// AutoregressiveDecoder_48619029791019
// MI455X (gfx1250) — compile-verified, matrix operand order checked by reading, not run on hardware
//
#include <hip/hip_runtime.h>

// AutoregressiveDecoder for MI455X (gfx1250), wave32 + WMMA f16->f32.
// N=256, D_IN=128, H1=256, H2=128.
// Restructure: B1 = z@W1[:128] once; conv_i(X) = d^2*X + d*m*(adjz@(m*d*X));
// mask/deg scaling folded into adjz COLUMNS (K-contiguous in A-operand layout);
// all GEMM B-operands read from pre-transposed buffers -> contiguous b128 loads.

#define NN   256
#define DIN  128
#define H1D  256
#define H2D  128
#define IB   16     // i-values per batch
#define NB   (NN / IB)

typedef __attribute__((ext_vector_type(16))) _Float16 v16h;
typedef __attribute__((ext_vector_type(8)))  float    v8f;

__device__ __forceinline__ v8f wmma16(v16h a, v16h b, v8f c) {
  // D = A(16x32 f16) x B(32x16 f16) + C(16x16 f32)
  return __builtin_amdgcn_wmma_f32_16x16x32_f16(false, a, false, b, (short)0, c, false, false);
}

// A-operand tile: rows r0..r0+15 of row-major f16 matrix M (ld halves).
// lane: m = r0 + lane%16 ; lanes 0-15 hold K=k0..k0+7 & k0+16..k0+23,
// lanes 16-31 hold K=k0+8..k0+15 & k0+24..k0+31.  (ISA 7.12.2 16-bit A layout)
// Identical layout serves the B operand when M is the transposed RHS (rows = columns).
__device__ __forceinline__ v16h ldA(const _Float16* M, int ld, int r0, int k0, int lane) {
  const _Float16* p = M + (r0 + (lane & 15)) * ld + k0 + ((lane >> 4) << 3);
  v16h a;
#pragma unroll
  for (int j = 0; j < 8; ++j) a[j] = p[j];
#pragma unroll
  for (int j = 0; j < 8; ++j) a[8 + j] = p[16 + j];
  return a;
}

// A-operand tile with per-COLUMN (K) f16 scale: computes rows of adjz*diag(rs).
// Contiguous loads; scaling is packed f16 VALU work. rs[k] in f16 (exact for 0/1 adjz).
__device__ __forceinline__ v16h ldA_colscaled(const _Float16* M, int ld, int r0, int k0,
                                              const _Float16* rs, int lane) {
  const _Float16* p = M + (r0 + (lane & 15)) * ld + k0 + ((lane >> 4) << 3);
  const _Float16* s = rs + k0 + ((lane >> 4) << 3);
  v16h a;
#pragma unroll
  for (int j = 0; j < 8; ++j) a[j] = p[j] * s[j];
#pragma unroll
  for (int j = 0; j < 8; ++j) a[8 + j] = p[16 + j] * s[16 + j];
  return a;
}

// ---------------- prep: f16 copies + transposes ----------------
__global__ void k_prep(const float* z, const float* adj, const float* W1, const float* W2,
                       _Float16* zh, _Float16* adjz, _Float16* W1T, _Float16* W2T,
                       float* w1last) {
  int t = blockIdx.x * blockDim.x + threadIdx.x;          // 0..65535
  if (t < NN * DIN) zh[t] = (_Float16)z[t];
  { int u = t >> 8, v = t & 255;                          // adj with zeroed diagonal
    adjz[t] = (u == v) ? (_Float16)0.0f : (_Float16)adj[t]; }
  if (t < H1D * DIN) {                                    // W1T[n*128+k] = W1[k*256+n], k<128
    int n = t >> 7, k = t & 127;
    W1T[t] = (_Float16)W1[k * H1D + n];
  }
  if (t < H2D * H1D) {                                    // W2T[n*256+k] = W2[k*128+n]
    int n = t >> 8, k = t & 255;
    W2T[t] = (_Float16)W2[k * H2D + n];
  }
  if (t < H1D) w1last[t] = W1[DIN * H1D + t];             // helper row of W1
}

// DEG[i][u] = (u<i) ? (1 + sum_{v<i, v!=u} adj[v][u])^-1/2 : 1   (column prefix sums)
__global__ void k_deg(const float* adj, float* DEG) {
  int u = threadIdx.x;
  float c = 0.0f;
  for (int i = 0; i < NN; ++i) {
    DEG[i * NN + u] = (u < i) ? rsqrtf(1.0f + c) : 1.0f;
    if (i != u) c += adj[i * NN + u];
  }
}

// ---------------- B1 = z @ W1[:128]  (f16 out, row-major + transposed) ----------------
__global__ void k_b1(const _Float16* zh, const _Float16* W1T, _Float16* B1h, _Float16* B1T) {
  int tid = threadIdx.x, lane = tid & 31, wave = tid >> 5;
  int n0 = blockIdx.x * 16;
  for (int mt = wave; mt < 16; mt += 8) {
    int m0 = mt * 16;
    v8f c = {};
    for (int k0 = 0; k0 < DIN; k0 += 32)
      c = wmma16(ldA(zh, DIN, m0, k0, lane), ldA(W1T, DIN, n0, k0, lane), c);
    int n = n0 + (lane & 15), mb = m0 + ((lane >> 4) << 3);
#pragma unroll
    for (int r = 0; r < 8; ++r) {
      _Float16 v = (_Float16)c[r];
      B1h[(mb + r) * H1D + n] = v;          // row-major (epilogue term)
      B1T[n * NN + (mb + r)]  = v;          // transposed (stage1 B operand)
    }
  }
}

// ---------------- stage1: H1 = relu(conv_i(B1 + e_i w1last)) ----------------
// Z = (adjz*diag(dm)) @ B1 ; B operand = rows of B1T (contiguous).
__global__ void k_stage1(const _Float16* adjz, const _Float16* B1h, const _Float16* B1T,
                         const float* w1last, const float* DEG, _Float16* H1buf, int i_base) {
  int ib = blockIdx.y, i = i_base + ib;
  __shared__ float    dd[NN];
  __shared__ _Float16 dmh[NN];
  for (int t = threadIdx.x; t < NN; t += blockDim.x) {
    float d = DEG[i * NN + t];
    dd[t]  = d;
    dmh[t] = (t < i) ? (_Float16)d : (_Float16)0.0f;  // mask*deg; row i masked -> helper excluded
  }
  __syncthreads();
  _Float16* H1 = H1buf + (size_t)ib * NN * H1D;
  int tid = threadIdx.x, lane = tid & 31, wave = tid >> 5;
  int n0 = blockIdx.x * 16;
  for (int mt = wave; mt < 16; mt += 8) {
    int m0 = mt * 16;
    v8f c = {};
    for (int k0 = 0; k0 < NN; k0 += 32) {
      if (k0 + 32 < NN) __builtin_prefetch(adjz + (m0 + (lane & 15)) * NN + k0 + 32, 0, 0);
      c = wmma16(ldA_colscaled(adjz, NN, m0, k0, dmh, lane),
                 ldA(B1T, NN, n0, k0, lane), c);
    }
    int n = n0 + (lane & 15), mb = m0 + ((lane >> 4) << 3);
#pragma unroll
    for (int r = 0; r < 8; ++r) {
      int u = mb + r;
      float x1 = (float)B1h[u * H1D + n] + ((u == i) ? w1last[n] : 0.0f);
      float d  = dd[u];
      float zt = (u < i) ? c[r] : 0.0f;           // m[u] * Z[u,n]
      float h  = d * (d * x1 + zt);
      H1[u * H1D + n] = (_Float16)fmaxf(h, 0.0f);
    }
  }
}

// ---------------- stage2: X2 = H1 @ W2 (f16 out, row-major + transposed) ----------------
__global__ void k_stage2(const _Float16* H1buf, const _Float16* W2T,
                         _Float16* X2buf, _Float16* X2Tbuf) {
  int ib = blockIdx.y;
  const _Float16* H1 = H1buf + (size_t)ib * NN * H1D;
  _Float16* X2  = X2buf  + (size_t)ib * NN * H2D;
  _Float16* X2T = X2Tbuf + (size_t)ib * NN * H2D;
  int tid = threadIdx.x, lane = tid & 31, wave = tid >> 5;
  int n0 = blockIdx.x * 16;                       // N = 128 -> gridDim.x = 8
  for (int mt = wave; mt < 16; mt += 8) {
    int m0 = mt * 16;
    v8f c = {};
    for (int k0 = 0; k0 < H1D; k0 += 32)
      c = wmma16(ldA(H1, H1D, m0, k0, lane), ldA(W2T, H1D, n0, k0, lane), c);
    int n = n0 + (lane & 15), mb = m0 + ((lane >> 4) << 3);
#pragma unroll
    for (int r = 0; r < 8; ++r) {
      _Float16 v = (_Float16)c[r];
      X2[(mb + r) * H2D + n] = v;           // row-major (stage3 epilogue)
      X2T[n * NN + (mb + r)] = v;           // transposed (stage3 B operand)
    }
  }
}

// ---------------- stage3: H2 = conv_i(X2) (f32 out) ----------------
__global__ void k_stage3(const _Float16* adjz, const _Float16* X2buf, const _Float16* X2Tbuf,
                         const float* DEG, float* H2buf, int i_base) {
  int ib = blockIdx.y, i = i_base + ib;
  __shared__ float    dd[NN];
  __shared__ _Float16 dmh[NN];
  for (int t = threadIdx.x; t < NN; t += blockDim.x) {
    float d = DEG[i * NN + t];
    dd[t]  = d;
    dmh[t] = (t < i) ? (_Float16)d : (_Float16)0.0f;
  }
  __syncthreads();
  const _Float16* X2  = X2buf  + (size_t)ib * NN * H2D;
  const _Float16* X2T = X2Tbuf + (size_t)ib * NN * H2D;
  float* H2 = H2buf + (size_t)ib * NN * H2D;
  int tid = threadIdx.x, lane = tid & 31, wave = tid >> 5;
  int n0 = blockIdx.x * 16;                       // N = 128 -> gridDim.x = 8
  for (int mt = wave; mt < 16; mt += 8) {
    int m0 = mt * 16;
    v8f c = {};
    for (int k0 = 0; k0 < NN; k0 += 32) {
      if (k0 + 32 < NN) __builtin_prefetch(adjz + (m0 + (lane & 15)) * NN + k0 + 32, 0, 0);
      c = wmma16(ldA_colscaled(adjz, NN, m0, k0, dmh, lane),
                 ldA(X2T, NN, n0, k0, lane), c);
    }
    int n = n0 + (lane & 15), mb = m0 + ((lane >> 4) << 3);
#pragma unroll
    for (int r = 0; r < 8; ++r) {
      int u = mb + r;
      float x2 = (float)X2[u * H2D + n];
      float d  = dd[u];
      float zt = (u < i) ? c[r] : 0.0f;
      H2[u * H2D + n] = d * (d * x2 + zt);
    }
  }
}

// ---------------- scores: supp[i][u] = (u<=i) ? 0.5*tanh(H2[u]·H2[i]) : 0 ----------------
__global__ void k_scores(const float* H2buf, float* supp, int i_base) {
  int ib = blockIdx.x, i = i_base + ib;
  const float* H2 = H2buf + (size_t)ib * NN * H2D;
  __shared__ float hi[H2D];
  for (int t = threadIdx.x; t < H2D; t += blockDim.x) hi[t] = H2[i * H2D + t];
  __syncthreads();
  int u = threadIdx.x;
  float acc = 0.0f;
#pragma unroll 4
  for (int k = 0; k < H2D; ++k) acc += H2[u * H2D + k] * hi[k];
  supp[i * NN + u] = (u <= i) ? 0.5f * tanhf(acc) : 0.0f;
}

// ---------------- final: out = 0.5*z@z^T + supp + supp^T ----------------
__global__ void k_final(const _Float16* zh, const float* supp, float* out) {
  int t = blockIdx.x;                              // 256 tiles, 1 wave each
  int mt = t >> 4, nt = t & 15;
  int lane = threadIdx.x;
  v8f c = {};
  for (int k0 = 0; k0 < DIN; k0 += 32)             // z@z^T: B operand = A-pattern rows of zh
    c = wmma16(ldA(zh, DIN, mt * 16, k0, lane), ldA(zh, DIN, nt * 16, k0, lane), c);
  int n = nt * 16 + (lane & 15), mb = mt * 16 + ((lane >> 4) << 3);
#pragma unroll
  for (int r = 0; r < 8; ++r) {
    int m = mb + r;
    out[m * NN + n] = 0.5f * c[r] + supp[m * NN + n] + supp[n * NN + m];
  }
}

extern "C" void kernel_launch(void* const* d_in, const int* in_sizes, int n_in,
                              void* d_out, int out_size, void* d_ws, size_t ws_size,
                              hipStream_t stream) {
  const float* z   = (const float*)d_in[0];   // [256,128]
  const float* adj = (const float*)d_in[1];   // [256,256]
  const float* W1  = (const float*)d_in[2];   // [129,256]
  const float* W2  = (const float*)d_in[3];   // [256,128]
  float* out = (float*)d_out;                 // [256,256]

  // carve workspace (256B aligned bumps), total ~7.4 MB
  char* base = (char*)d_ws;
  size_t off = 0;
  auto bump = [&](size_t bytes) {
    void* p = base + off;
    off += (bytes + 255) & ~(size_t)255;
    return p;
  };
  _Float16* zh     = (_Float16*)bump((size_t)NN * DIN * 2);
  _Float16* adjz   = (_Float16*)bump((size_t)NN * NN * 2);
  _Float16* W1T    = (_Float16*)bump((size_t)H1D * DIN * 2);
  _Float16* W2T    = (_Float16*)bump((size_t)H2D * H1D * 2);
  float*    w1last = (float*)bump((size_t)H1D * 4);
  float*    DEG    = (float*)bump((size_t)NN * NN * 4);
  float*    supp   = (float*)bump((size_t)NN * NN * 4);
  _Float16* B1h    = (_Float16*)bump((size_t)NN * H1D * 2);
  _Float16* B1T    = (_Float16*)bump((size_t)NN * H1D * 2);
  _Float16* H1buf  = (_Float16*)bump((size_t)IB * NN * H1D * 2);
  _Float16* X2buf  = (_Float16*)bump((size_t)IB * NN * H2D * 2);
  _Float16* X2Tbuf = (_Float16*)bump((size_t)IB * NN * H2D * 2);
  float*    H2buf  = (float*)bump((size_t)IB * NN * H2D * 4);
  (void)ws_size; (void)in_sizes; (void)n_in; (void)out_size;

  k_prep<<<dim3(NN), dim3(256), 0, stream>>>(z, adj, W1, W2, zh, adjz, W1T, W2T, w1last);
  k_deg<<<dim3(1), dim3(NN), 0, stream>>>(adj, DEG);
  k_b1<<<dim3(16), dim3(256), 0, stream>>>(zh, W1T, B1h, B1T);

  for (int b = 0; b < NB; ++b) {
    int i_base = b * IB;
    k_stage1<<<dim3(16, IB), dim3(256), 0, stream>>>(adjz, B1h, B1T, w1last, DEG, H1buf, i_base);
    k_stage2<<<dim3(8, IB),  dim3(256), 0, stream>>>(H1buf, W2T, X2buf, X2Tbuf);
    k_stage3<<<dim3(8, IB),  dim3(256), 0, stream>>>(adjz, X2buf, X2Tbuf, DEG, H2buf, i_base);
    k_scores<<<dim3(IB),     dim3(256), 0, stream>>>(H2buf, supp, i_base);
  }
  k_final<<<dim3(256), dim3(32), 0, stream>>>(zh, supp, out);
}
